// Decoder_1116691497812
// MI455X (gfx1250) — matrix operand order repaired and checked by reading, not compiled, not run on hardware
//
#include <hip/hip_runtime.h>
#include <hip/hip_bf16.h>

typedef __attribute__((ext_vector_type(16))) __bf16 v16bf;
typedef __attribute__((ext_vector_type(8)))  __bf16 v8bf;
typedef __attribute__((ext_vector_type(4)))  __bf16 v4bf;
typedef __attribute__((ext_vector_type(8)))  float  v8f;

#define B_    256
#define L_    196
#define FEAT_ 2048
#define HID_  1024
#define EMB_  512
#define VOCAB_ 10000
#define ATT_  1024
#define ML_   (B_ * L_)      // 50176
#define XDIM_ (EMB_ + FEAT_) // 2560

// ---------------- WMMA tile loaders -----------------
// A tile (16x32, 16-bit): lane<16 -> M=lane, K = e<8 ? kb+e : kb+16+(e&7)
//                         lane>=16 -> same M=lane&15, K runs shifted by +8.
__device__ inline v16bf ld_a_g(const float* __restrict__ A, int lda, int row,
                               int kb, int hi) {
  const float* p = A + (size_t)row * lda + kb + 8 * hi;
  float4 r0 = *(const float4*)(p);
  float4 r1 = *(const float4*)(p + 4);
  float4 r2 = *(const float4*)(p + 16);
  float4 r3 = *(const float4*)(p + 20);
  v16bf a;
  a[0]=(__bf16)r0.x;  a[1]=(__bf16)r0.y;  a[2]=(__bf16)r0.z;  a[3]=(__bf16)r0.w;
  a[4]=(__bf16)r1.x;  a[5]=(__bf16)r1.y;  a[6]=(__bf16)r1.z;  a[7]=(__bf16)r1.w;
  a[8]=(__bf16)r2.x;  a[9]=(__bf16)r2.y;  a[10]=(__bf16)r2.z; a[11]=(__bf16)r2.w;
  a[12]=(__bf16)r3.x; a[13]=(__bf16)r3.y; a[14]=(__bf16)r3.z; a[15]=(__bf16)r3.w;
  return a;
}

// B tile (32x16, 16-bit) from row-major f32 W[N][K]:
// lane n holds K = e<8 ? kb+8*hi+e : kb+16+8*hi+(e&7)  (same order as A).
__device__ inline v16bf ld_b_g(const float* __restrict__ W, int ldw, int n,
                               int kb, int hi) {
  const float* p = W + (size_t)n * ldw + kb + 8 * hi;
  float4 r0 = *(const float4*)(p);
  float4 r1 = *(const float4*)(p + 4);
  float4 r2 = *(const float4*)(p + 16);
  float4 r3 = *(const float4*)(p + 20);
  v16bf b;
  b[0]=(__bf16)r0.x;  b[1]=(__bf16)r0.y;  b[2]=(__bf16)r0.z;  b[3]=(__bf16)r0.w;
  b[4]=(__bf16)r1.x;  b[5]=(__bf16)r1.y;  b[6]=(__bf16)r1.z;  b[7]=(__bf16)r1.w;
  b[8]=(__bf16)r2.x;  b[9]=(__bf16)r2.y;  b[10]=(__bf16)r2.z; b[11]=(__bf16)r2.w;
  b[12]=(__bf16)r3.x; b[13]=(__bf16)r3.y; b[14]=(__bf16)r3.z; b[15]=(__bf16)r3.w;
  return b;
}

// B tile from pre-converted bf16 W[N][K]: two 16-byte loads, no conversion.
__device__ inline v16bf ld_b_bf(const __bf16* __restrict__ W, int ldw, int n,
                                int kb, int hi) {
  const __bf16* p = W + (size_t)n * ldw + kb + 8 * hi;
  v8bf lo = *(const v8bf*)(p);
  v8bf hh = *(const v8bf*)(p + 16);
  v16bf b;
#pragma unroll
  for (int e = 0; e < 8; e++) { b[e] = lo[e]; b[8 + e] = hh[e]; }
  return b;
}

__device__ inline v8f wmma_bf16(v16bf a, v16bf b, v8f c) {
  return __builtin_amdgcn_wmma_f32_16x16x32_bf16(false, a, false, b, (short)0,
                                                 c, false, false);
}

// K-loop GEMM accumulate: C(16x16) += A[mbase: , :K] * W[nbase: , :K]^T
__device__ inline void gemm_acc(v8f& acc, const float* __restrict__ A, int lda,
                                const float* __restrict__ W, int ldw,
                                int mbase, int nbase, int K, int lr, int hi) {
#pragma unroll 2
  for (int kb = 0; kb < K; kb += 32) {
    v16bf a = ld_a_g(A, lda, mbase + lr, kb, hi);
    v16bf b = ld_b_g(W, ldw, nbase + lr, kb, hi);
    acc = wmma_bf16(a, b, acc);
  }
}

// ---------------- prep kernels ----------------
__global__ void k_cvt_bf16(const float* __restrict__ src,
                           __bf16* __restrict__ dst, int n4) {
  int i = blockIdx.x * 256 + threadIdx.x;
  if (i < n4) {
    float4 v = ((const float4*)src)[i];
    v4bf w;
    w[0] = (__bf16)v.x; w[1] = (__bf16)v.y;
    w[2] = (__bf16)v.z; w[3] = (__bf16)v.w;
    ((v4bf*)dst)[i] = w;
  }
}

__global__ void k_embed(const int* __restrict__ token,
                        const float* __restrict__ emb, float* __restrict__ x) {
  int idx = blockIdx.x * 256 + threadIdx.x;  // 256*512
  int b = idx >> 9, e = idx & 511;
  x[(size_t)b * XDIM_ + e] = emb[(size_t)token[b] * EMB_ + e];
}

// hpb[b,a] = hidden[b,:]·Wh[a,:] + bf[a] + bh[a]
__global__ void k_hidproj(const float* __restrict__ hidden,
                          const float* __restrict__ Wh,
                          const float* __restrict__ bf,
                          const float* __restrict__ bh,
                          float* __restrict__ hpb) {
  int tile = blockIdx.x * 4 + (threadIdx.x >> 5);  // 1024 tiles: 16 m x 64 n
  int lane = threadIdx.x & 31, lr = lane & 15, hi = lane >> 4;
  int mbase = (tile & 15) * 16, nbase = (tile >> 4) * 16;
  v8f acc = {};
  gemm_acc(acc, hidden, HID_, Wh, HID_, mbase, nbase, HID_, lr, hi);
  int col = nbase + lr;
  float bias = bf[col] + bh[col];
#pragma unroll
  for (int j = 0; j < 8; j++)
    hpb[(size_t)(mbase + j + 8 * hi) * ATT_ + col] = acc[j] + bias;
}

// ---------------- big attention-score kernel ----------------
// score_part[wave][m] = sum over wave's N-quarter of relu(feat·Wf + hpb)·Wa
__global__ void k_attn(const float* __restrict__ features,
                       const __bf16* __restrict__ Wf,   // pre-converted bf16
                       const float* __restrict__ Wa,
                       const float* __restrict__ hpb,
                       float* __restrict__ score_part) {
  extern __shared__ __bf16 As[];  // 64 rows x 2048 K, bf16 = 256KB
  const int wid = threadIdx.x >> 5;
  const int lane = threadIdx.x & 31, lr = lane & 15, hi = lane >> 4;
  const int wgbase = blockIdx.x * 64;

  // Stage 64 feature rows into LDS as bf16 (features read from HBM once).
  {
    const float4* src = (const float4*)(features + (size_t)wgbase * FEAT_);
    for (int i = threadIdx.x; i < 64 * FEAT_ / 4; i += 128) {
      float4 v = src[i];
      v4bf w;
      w[0] = (__bf16)v.x; w[1] = (__bf16)v.y;
      w[2] = (__bf16)v.z; w[3] = (__bf16)v.w;
      *(v4bf*)(As + (size_t)i * 4) = w;
    }
  }
  __syncthreads();

  float part[4][8];
#pragma unroll
  for (int rb = 0; rb < 4; rb++)
#pragma unroll
    for (int j = 0; j < 8; j++) part[rb][j] = 0.f;

  for (int chunk = 0; chunk < 16; ++chunk) {
    const int nbase = wid * 256 + chunk * 16;  // this wave's N quarter
    const float wa = Wa[nbase + lr];
    v8f acc[4];
#pragma unroll
    for (int rb = 0; rb < 4; rb++) acc[rb] = (v8f){};

#pragma unroll 2
    for (int kb = 0; kb < FEAT_; kb += 32) {
      v16bf b = ld_b_bf(Wf, FEAT_, nbase + lr, kb, hi);  // 2 loads, no cvt
#pragma unroll
      for (int rb = 0; rb < 4; rb++) {  // 4x A reuse per B load
        const __bf16* ap = As + (size_t)(rb * 16 + lr) * FEAT_ + kb + 8 * hi;
        v8bf lo = *(const v8bf*)(ap);
        v8bf hh = *(const v8bf*)(ap + 16);
        v16bf a;
#pragma unroll
        for (int e = 0; e < 8; e++) { a[e] = lo[e]; a[8 + e] = hh[e]; }
        acc[rb] = wmma_bf16(a, b, acc[rb]);
      }
    }
    // fuse: + hid_proj, relu, ·Wa, accumulate per-row partial score
#pragma unroll
    for (int rb = 0; rb < 4; rb++) {
#pragma unroll
      for (int j = 0; j < 8; j++) {
        int m = wgbase + rb * 16 + j + 8 * hi;
        unsigned bidx = (unsigned)m / L_;
        float v = acc[rb][j] + hpb[(size_t)bidx * ATT_ + nbase + lr];
        part[rb][j] += fmaxf(v, 0.f) * wa;
      }
    }
  }
  // reduce across the 16 N-lanes of each half-wave; write per-wave partial
#pragma unroll
  for (int rb = 0; rb < 4; rb++) {
#pragma unroll
    for (int j = 0; j < 8; j++) {
      float v = part[rb][j];
      v += __shfl_xor(v, 1, 32);
      v += __shfl_xor(v, 2, 32);
      v += __shfl_xor(v, 4, 32);
      v += __shfl_xor(v, 8, 32);
      if (lr == 0)
        score_part[(size_t)wid * ML_ + wgbase + rb * 16 + j + 8 * hi] = v;
    }
  }
}

// softmax over L per batch (ba dropped: softmax shift-invariant) + beta gate
__global__ void k_softmax_beta(const float* __restrict__ score_part,
                               const float* __restrict__ hidden,
                               const float* __restrict__ sel_W,
                               const float* __restrict__ sel_b,
                               float* __restrict__ alpha,
                               float* __restrict__ beta) {
  __shared__ float red[256];
  int b = blockIdx.x, t = threadIdx.x;
  float s = -1e30f;
  if (t < L_) {
    size_t i = (size_t)b * L_ + t;
    s = score_part[i] + score_part[ML_ + i] + score_part[2 * ML_ + i] +
        score_part[3 * ML_ + i];
  }
  red[t] = s;
  __syncthreads();
  for (int o = 128; o > 0; o >>= 1) {
    if (t < o) red[t] = fmaxf(red[t], red[t + o]);
    __syncthreads();
  }
  float mx = red[0];
  __syncthreads();
  float e = (t < L_) ? __expf(s - mx) : 0.f;
  red[t] = e;
  __syncthreads();
  for (int o = 128; o > 0; o >>= 1) {
    if (t < o) red[t] += red[t + o];
    __syncthreads();
  }
  float inv = 1.f / red[0];
  if (t < L_) alpha[(size_t)b * L_ + t] = e * inv;
  __syncthreads();
  float p = 0.f;
  for (int k = t; k < HID_; k += 256) p += hidden[(size_t)b * HID_ + k] * sel_W[k];
  red[t] = p;
  __syncthreads();
  for (int o = 128; o > 0; o >>= 1) {
    if (t < o) red[t] += red[t + o];
    __syncthreads();
  }
  if (t == 0) beta[b] = 1.f / (1.f + __expf(-(red[0] + sel_b[0])));
}

// context[b,f] = beta[b]/L * sum_l alpha[b,l]*features[b,l,f] -> x[:,512+f]
__global__ void k_context(const float* __restrict__ features,
                          const float* __restrict__ alpha,
                          const float* __restrict__ beta,
                          float* __restrict__ x) {
  __shared__ float al[L_];
  int b = blockIdx.x >> 3, fc = blockIdx.x & 7;
  int f = fc * 256 + threadIdx.x;
  for (int l = threadIdx.x; l < L_; l += 256) al[l] = alpha[(size_t)b * L_ + l];
  __syncthreads();
  const float* fp = features + (size_t)b * L_ * FEAT_ + f;
  float acc = 0.f;
#pragma unroll 4
  for (int l = 0; l < L_; l++) acc += al[l] * fp[(size_t)l * FEAT_];
  x[(size_t)b * XDIM_ + EMB_ + f] = acc * beta[b] * (1.f / (float)L_);
}

// gates = x·Wih^T + hidden·Whh^T + bih + bhh    (M=256, N=4096)
__global__ void k_gates(const float* __restrict__ x,
                        const float* __restrict__ hidden,
                        const float* __restrict__ Wih,
                        const float* __restrict__ Whh,
                        const float* __restrict__ bih,
                        const float* __restrict__ bhh,
                        float* __restrict__ gates) {
  int tile = blockIdx.x * 4 + (threadIdx.x >> 5);  // 4096 tiles: 16 m x 256 n
  int lane = threadIdx.x & 31, lr = lane & 15, hi = lane >> 4;
  int mbase = (tile & 15) * 16, nbase = (tile >> 4) * 16;
  v8f acc = {};
  gemm_acc(acc, x, XDIM_, Wih, XDIM_, mbase, nbase, XDIM_, lr, hi);
  gemm_acc(acc, hidden, HID_, Whh, HID_, mbase, nbase, HID_, lr, hi);
  int col = nbase + lr;
  float bias = bih[col] + bhh[col];
#pragma unroll
  for (int j = 0; j < 8; j++)
    gates[(size_t)(mbase + j + 8 * hi) * (4 * HID_) + col] = acc[j] + bias;
}

__global__ void k_lstm(const float* __restrict__ gates,
                       const float* __restrict__ cell,
                       float* __restrict__ new_c, float* __restrict__ new_h) {
  int idx = blockIdx.x * 256 + threadIdx.x;  // 256*1024
  int b = idx >> 10, h = idx & 1023;
  const float* g = gates + (size_t)b * 4 * HID_;
  float si = 1.f / (1.f + __expf(-g[h]));
  float sf = 1.f / (1.f + __expf(-g[HID_ + h]));
  float gg = tanhf(g[2 * HID_ + h]);
  float so = 1.f / (1.f + __expf(-g[3 * HID_ + h]));
  float nc = sf * cell[idx] + si * gg;
  new_c[idx] = nc;
  new_h[idx] = so * tanhf(nc);
}

// logit = new_h·deepW^T + deep_b   (M=256, N=10000, K=1024)
__global__ void k_logit(const float* __restrict__ newh,
                        const float* __restrict__ Wd,
                        const float* __restrict__ bd,
                        float* __restrict__ logit) {
  int tile = blockIdx.x * 4 + (threadIdx.x >> 5);  // 10000 tiles: 16 m x 625 n
  int lane = threadIdx.x & 31, lr = lane & 15, hi = lane >> 4;
  int mbase = ((unsigned)tile % 16u) * 16, nbase = ((unsigned)tile / 16u) * 16;
  v8f acc = {};
  gemm_acc(acc, newh, HID_, Wd, HID_, mbase, nbase, HID_, lr, hi);
  int col = nbase + lr;
  float bias = bd[col];
#pragma unroll
  for (int j = 0; j < 8; j++)
    logit[(size_t)(mbase + j + 8 * hi) * VOCAB_ + col] = acc[j] + bias;
}

// ---------------- launcher ----------------
extern "C" void kernel_launch(void* const* d_in, const int* in_sizes, int n_in,
                              void* d_out, int out_size, void* d_ws,
                              size_t ws_size, hipStream_t stream) {
  const int*   token     = (const int*)d_in[0];
  const float* features  = (const float*)d_in[1];
  const float* hidden    = (const float*)d_in[2];
  const float* cell      = (const float*)d_in[3];
  const float* emb_table = (const float*)d_in[4];
  const float* att_Wf    = (const float*)d_in[5];
  const float* att_bf    = (const float*)d_in[6];
  const float* att_Wh    = (const float*)d_in[7];
  const float* att_bh    = (const float*)d_in[8];
  const float* att_Wa    = (const float*)d_in[9];
  // d_in[10] = att_ba: dropped (softmax shift-invariant)
  const float* sel_W     = (const float*)d_in[11];
  const float* sel_b     = (const float*)d_in[12];
  const float* W_ih      = (const float*)d_in[13];
  const float* b_ih      = (const float*)d_in[14];
  const float* W_hh      = (const float*)d_in[15];
  const float* b_hh      = (const float*)d_in[16];
  const float* deep_W    = (const float*)d_in[17];
  const float* deep_b    = (const float*)d_in[18];

  float* out   = (float*)d_out;
  float* logit = out;                                   // 256*10000
  float* new_h = out + (size_t)B_ * VOCAB_;             // 256*1024
  float* new_c = new_h + (size_t)B_ * HID_;             // 256*1024
  float* alpha = new_c + (size_t)B_ * HID_;             // 256*196

  float*  ws    = (float*)d_ws;
  float*  score = ws;                                   // 4 * 50176
  float*  hpb   = score + 4 * ML_;                      // 256*1024
  float*  beta  = hpb + (size_t)B_ * ATT_;              // 256
  float*  xbuf  = beta + 256;                           // 256*2560
  float*  gates = xbuf + (size_t)B_ * XDIM_;            // 256*4096
  __bf16* wf_bf = (__bf16*)(gates + (size_t)B_ * 4 * HID_);  // 1024*2048 bf16

  // pre-convert att_Wf to bf16 (hot GEMM's B stream: 2 loads/tile, no cvt)
  k_cvt_bf16<<<(ATT_ * FEAT_ / 4 + 255) / 256, 256, 0, stream>>>(
      att_Wf, wf_bf, ATT_ * FEAT_ / 4);
  k_embed<<<(B_ * EMB_) / 256, 256, 0, stream>>>(token, emb_table, xbuf);
  k_hidproj<<<256, 128, 0, stream>>>(hidden, att_Wh, att_bf, att_bh, hpb);
  k_attn<<<ML_ / 64, 128, 64 * FEAT_ * sizeof(__bf16), stream>>>(
      features, wf_bf, att_Wa, hpb, score);
  k_softmax_beta<<<B_, 256, 0, stream>>>(score, hidden, sel_W, sel_b, alpha,
                                         beta);
  k_context<<<B_ * 8, 256, 0, stream>>>(features, alpha, beta, xbuf);
  k_gates<<<1024, 128, 0, stream>>>(xbuf, hidden, W_ih, W_hh, b_ih, b_hh,
                                    gates);
  k_lstm<<<(B_ * HID_) / 256, 256, 0, stream>>>(gates, cell, new_c, new_h);
  k_logit<<<2500, 128, 0, stream>>>(new_h, deep_W, deep_b, logit);
}
